// LSTM_8100308320531
// MI455X (gfx1250) — matrix operand order repaired and checked by reading, not compiled, not run on hardware
//
#include <hip/hip_runtime.h>
#include <hip/hip_bf16.h>

// ---------------------------------------------------------------------------
// 2-layer LSTM (B=512, T=300, D=80, H=128) + MLP head for MI455X (gfx1250).
// Strategy: persistent per-batch-tile workgroups; recurrence GEMMs on
// v_wmma_f32_16x16x32_f16 with weight fragments pinned in VGPRs across all
// 300 time steps; h/c state in LDS/registers; no device-wide sync.
// ---------------------------------------------------------------------------

typedef __attribute__((ext_vector_type(16))) _Float16 v16h;
typedef __attribute__((ext_vector_type(8)))  _Float16 v8h;
typedef __attribute__((ext_vector_type(8)))  float    v8f;

#define T_STEPS 300
#define WMMA_F16(a, b, c) \
    __builtin_amdgcn_wmma_f32_16x16x32_f16(false, (a), false, (b), (short)0, (c), false, false)

__device__ __forceinline__ float sig_(float x)  { return 1.0f / (1.0f + __expf(-x)); }
__device__ __forceinline__ float tanh_(float x) { return 1.0f - 2.0f / (__expf(2.0f * x) + 1.0f); }

// A-matrix (16xK f16, row-major, row stride ldk halves) fragment for one
// K-block of 32.  Contiguous layout: lanes 0-15 = rows, halves 0..15 ->
// K=k0..k0+15; lanes 16-31 take K=k0+16..k0+31 (matches load_b_frag).
__device__ __forceinline__ v16h load_a_frag(const _Float16* base, int ldk, int k0, int lane) {
    const int row = lane & 15;
    const _Float16* p = base + row * ldk + k0 + ((lane & 16) ? 16 : 0);
    v8h lo = *(const v8h*)(p);
    v8h hi = *(const v8h*)(p + 8);
    v16h r;
#pragma unroll
    for (int i = 0; i < 8; ++i) { r[i] = lo[i]; r[i + 8] = hi[i]; }
    return r;
}

// B-matrix (32xK x 16N) fragment from row-major W[N][K] f16: lane n holds 16
// contiguous K values of column n0+(lane&15); lanes 16-31 take K=k0+16..k0+31.
// -> a single 32B contiguous load per lane.
__device__ __forceinline__ v16h load_b_frag(const _Float16* W, int ldk, int n0, int k0, int lane) {
    const int n = n0 + (lane & 15);
    const int k = k0 + ((lane & 16) ? 16 : 0);
    return *(const v16h*)(W + (size_t)n * ldk + k);
}

// ---------------------------------------------------------------------------
// Prep: x (B,D,T) f32 -> xT (T,B,96) f16, K padded 80->96 with zeros.
// ---------------------------------------------------------------------------
__global__ __launch_bounds__(256) void prep_x(const float* __restrict__ x,
                                              _Float16* __restrict__ xT) {
    const int idx = blockIdx.x * 256 + threadIdx.x;
    const int total = T_STEPS * 512 * 96;
    if (idx >= total) return;
    const int t = idx / (512 * 96);
    const int r = idx - t * 512 * 96;
    const int b = r / 96;
    const int d = r - b * 96;
    const float v = (d < 80) ? x[((size_t)b * 80 + d) * T_STEPS + t] : 0.0f;
    xT[idx] = (_Float16)v;
}

// ---------------------------------------------------------------------------
// Prep: weights -> f16 (W_ih0 K-padded 80->96), bias sums.
// ---------------------------------------------------------------------------
__global__ __launch_bounds__(256) void prep_w(
    const float* __restrict__ Wih0, const float* __restrict__ Whh0,
    const float* __restrict__ bih0, const float* __restrict__ bhh0,
    const float* __restrict__ Wih1, const float* __restrict__ Whh1,
    const float* __restrict__ bih1, const float* __restrict__ bhh1,
    _Float16* __restrict__ Wih0h, _Float16* __restrict__ Whh0h,
    _Float16* __restrict__ Wih1h, _Float16* __restrict__ Whh1h,
    float* __restrict__ b0, float* __restrict__ b1) {
    int idx = blockIdx.x * 256 + threadIdx.x;
    if (idx < 512 * 96) {
        const int g = idx / 96, d = idx - g * 96;
        Wih0h[idx] = (_Float16)((d < 80) ? Wih0[g * 80 + d] : 0.0f);
        return;
    }
    idx -= 512 * 96;
    if (idx < 512 * 128) { Whh0h[idx] = (_Float16)Whh0[idx]; return; }
    idx -= 512 * 128;
    if (idx < 512 * 128) { Wih1h[idx] = (_Float16)Wih1[idx]; return; }
    idx -= 512 * 128;
    if (idx < 512 * 128) { Whh1h[idx] = (_Float16)Whh1[idx]; return; }
    idx -= 512 * 128;
    if (idx < 512) { b0[idx] = bih0[idx] + bhh0[idx]; return; }
    idx -= 512;
    if (idx < 512) { b1[idx] = bih1[idx] + bhh1[idx]; return; }
}

// ---------------------------------------------------------------------------
// Persistent LSTM layer. One WG owns a 16-row batch tile for all 300 steps.
// KIN = padded input width (96 for layer 0, 128 for layer 1).
// STORE_SEQ: store h every step (layer 0) vs only the final h (layer 1).
// 8 waves; wave w computes gate columns {q*128 + w*16 .. +15} for q=i,f,g,o,
// so the cell update and persistent c-state are wave-local in registers.
// ---------------------------------------------------------------------------
template <int KIN, bool STORE_SEQ>
__global__ __launch_bounds__(256) void lstm_layer(
    const _Float16* __restrict__ xin,  // [T][512][KIN] time-major f16
    const _Float16* __restrict__ Wih,  // [512][KIN]
    const _Float16* __restrict__ Whh,  // [512][128]
    const float*    __restrict__ bsum, // [512] (b_ih + b_hh)
    _Float16*       __restrict__ hout) // STORE_SEQ ? [T][512][128] : [512][128]
{
    constexpr int NB_IN = KIN / 32;
    __shared__ _Float16 hbuf[16 * 128];
    __shared__ _Float16 xbuf[2][16 * KIN];

    const int tid   = threadIdx.x;
    const int lane  = tid & 31;
    const int w     = tid >> 5;           // wave 0..7
    const int bbase = blockIdx.x * 16;    // batch-tile origin

    // h0 = 0
    for (int i = tid; i < 16 * 128; i += 256) hbuf[i] = (_Float16)0;

    // Stage x tile for t=0 (flat 32*KIN-byte copy; src/dst layouts identical).
    {
        const uint4* src = (const uint4*)(xin + (size_t)bbase * KIN);
        if (tid < 2 * KIN) ((uint4*)xbuf[0])[tid] = src[tid];
    }

    // Pin all weight fragments for this wave's 4 gate tiles in VGPRs.
    v16h bIn[4][NB_IN];
    v16h bHh[4][4];
#pragma unroll
    for (int q = 0; q < 4; ++q) {
        const int n0 = q * 128 + w * 16;
#pragma unroll
        for (int kb = 0; kb < NB_IN; ++kb) bIn[q][kb] = load_b_frag(Wih, KIN, n0, kb * 32, lane);
#pragma unroll
        for (int kb = 0; kb < 4; ++kb)     bHh[q][kb] = load_b_frag(Whh, 128, n0, kb * 32, lane);
    }

    // Per-lane gate biases (C/D layout: column = lane&15 for both lane halves).
    float bias[4];
#pragma unroll
    for (int q = 0; q < 4; ++q) bias[q] = bsum[q * 128 + w * 16 + (lane & 15)];

    // Persistent cell state, C/D register layout (v8f: row r + 8*(lane>=16)).
    float cst[8];
#pragma unroll
    for (int r = 0; r < 8; ++r) cst[r] = 0.0f;

    const int col   = w * 16 + (lane & 15);
    const int rbase = (lane & 16) ? 8 : 0;

    __syncthreads();

    for (int t = 0; t < T_STEPS; ++t) {
        const _Float16* xcur = xbuf[t & 1];

        // gates = bias + x_t @ W_ih^T + h @ W_hh^T   (f32 accumulate)
        v8f acc[4];
#pragma unroll
        for (int q = 0; q < 4; ++q)
#pragma unroll
            for (int r = 0; r < 8; ++r) acc[q][r] = bias[q];

#pragma unroll
        for (int kb = 0; kb < NB_IN; ++kb) {
            v16h a = load_a_frag(xcur, KIN, kb * 32, lane);
#pragma unroll
            for (int q = 0; q < 4; ++q) acc[q] = WMMA_F16(a, bIn[q][kb], acc[q]);
        }
#pragma unroll
        for (int kb = 0; kb < 4; ++kb) {
            v16h a = load_a_frag(hbuf, 128, kb * 32, lane);
#pragma unroll
            for (int q = 0; q < 4; ++q) acc[q] = WMMA_F16(a, bHh[q][kb], acc[q]);
        }

        // Cell update, entirely in registers (i,f,g,o tiles share this wave).
        _Float16 hn[8];
#pragma unroll
        for (int r = 0; r < 8; ++r) {
            const float iv = sig_(acc[0][r]);
            const float fv = sig_(acc[1][r]);
            const float gv = tanh_(acc[2][r]);
            const float ov = sig_(acc[3][r]);
            const float cn = fv * cst[r] + iv * gv;
            cst[r] = cn;
            hn[r] = (_Float16)(ov * tanh_(cn));
        }

        __syncthreads();  // all waves finished reading old h

#pragma unroll
        for (int r = 0; r < 8; ++r) hbuf[(rbase + r) * 128 + col] = hn[r];

        // Stage next x tile into the other buffer; prefetch the one after.
        if (t + 1 < T_STEPS) {
            const uint4* src = (const uint4*)(xin + ((size_t)(t + 1) * 512 + bbase) * KIN);
            if (tid < 2 * KIN) ((uint4*)xbuf[(t + 1) & 1])[tid] = src[tid];
            if (t + 2 < T_STEPS && tid < KIN / 4)
                __builtin_prefetch(xin + ((size_t)(t + 2) * 512 + bbase) * KIN + (size_t)tid * 64, 0, 1);
        }

        __syncthreads();  // new h complete

        if (STORE_SEQ) {
            ((uint4*)(hout + ((size_t)t * 512 + bbase) * 128))[tid] = ((const uint4*)hbuf)[tid];
        } else if (t == T_STEPS - 1) {
            ((uint4*)(hout + (size_t)bbase * 128))[tid] = ((const uint4*)hbuf)[tid];
        }
    }
}

// ---------------------------------------------------------------------------
// MLP head: 128 -> 64 -> 32 -> 2 on the final hidden state. Tiny; plain VALU.
// ---------------------------------------------------------------------------
__global__ __launch_bounds__(256) void lstm_head(
    const _Float16* __restrict__ h2,
    const float* __restrict__ fc1w, const float* __restrict__ fc1b,
    const float* __restrict__ fc2w, const float* __restrict__ fc2b,
    const float* __restrict__ fc3w, const float* __restrict__ fc3b,
    float* __restrict__ out) {
    const int b = blockIdx.x * blockDim.x + threadIdx.x;
    if (b >= 512) return;
    const _Float16* h = h2 + (size_t)b * 128;
    float a1[64];
    for (int j = 0; j < 64; ++j) {
        float s = fc1b[j];
        const float* wr = fc1w + j * 128;
        for (int i = 0; i < 128; ++i) s += (float)h[i] * wr[i];
        a1[j] = fmaxf(s, 0.0f);
    }
    float a2[32];
    for (int j = 0; j < 32; ++j) {
        float s = fc2b[j];
        const float* wr = fc2w + j * 64;
        for (int i = 0; i < 64; ++i) s += a1[i] * wr[i];
        a2[j] = fmaxf(s, 0.0f);
    }
    for (int j = 0; j < 2; ++j) {
        float s = fc3b[j];
        const float* wr = fc3w + j * 32;
        for (int i = 0; i < 32; ++i) s += a2[i] * wr[i];
        out[b * 2 + j] = s;
    }
}

// ---------------------------------------------------------------------------
extern "C" void kernel_launch(void* const* d_in, const int* in_sizes, int n_in,
                              void* d_out, int out_size, void* d_ws, size_t ws_size,
                              hipStream_t stream) {
    (void)in_sizes; (void)n_in; (void)out_size; (void)ws_size;

    const float* x     = (const float*)d_in[0];
    const float* Wih0  = (const float*)d_in[1];
    const float* Whh0  = (const float*)d_in[2];
    const float* bih0  = (const float*)d_in[3];
    const float* bhh0  = (const float*)d_in[4];
    const float* Wih1  = (const float*)d_in[5];
    const float* Whh1  = (const float*)d_in[6];
    const float* bih1  = (const float*)d_in[7];
    const float* bhh1  = (const float*)d_in[8];
    const float* fc1w  = (const float*)d_in[9];
    const float* fc1b  = (const float*)d_in[10];
    const float* fc2w  = (const float*)d_in[11];
    const float* fc2b  = (const float*)d_in[12];
    const float* fc3w  = (const float*)d_in[13];
    const float* fc3b  = (const float*)d_in[14];

    char* ws = (char*)d_ws;
    size_t off = 0;
    auto alloc = [&](size_t bytes) {
        size_t o = off;
        off = (off + bytes + 255) & ~(size_t)255;
        return o;
    };
    _Float16* xT     = (_Float16*)(ws + alloc((size_t)T_STEPS * 512 * 96 * 2));   // 29.5 MB
    _Float16* Wih0h  = (_Float16*)(ws + alloc((size_t)512 * 96 * 2));
    _Float16* Whh0h  = (_Float16*)(ws + alloc((size_t)512 * 128 * 2));
    _Float16* Wih1h  = (_Float16*)(ws + alloc((size_t)512 * 128 * 2));
    _Float16* Whh1h  = (_Float16*)(ws + alloc((size_t)512 * 128 * 2));
    float*    b0     = (float*)   (ws + alloc((size_t)512 * 4));
    float*    b1     = (float*)   (ws + alloc((size_t)512 * 4));
    _Float16* h1     = (_Float16*)(ws + alloc((size_t)T_STEPS * 512 * 128 * 2));  // 39.3 MB
    _Float16* h2last = (_Float16*)(ws + alloc((size_t)512 * 128 * 2));

    // 1) x -> time-major f16 (K padded to 96)
    prep_x<<<(T_STEPS * 512 * 96 + 255) / 256, 256, 0, stream>>>(x, xT);
    // 2) weights -> f16, bias sums
    prep_w<<<(512 * 96 + 3 * 512 * 128 + 1024 + 255) / 256, 256, 0, stream>>>(
        Wih0, Whh0, bih0, bhh0, Wih1, Whh1, bih1, bhh1,
        Wih0h, Whh0h, Wih1h, Whh1h, b0, b1);
    // 3) layer 0: persistent recurrence, stores full h sequence
    lstm_layer<96, true><<<32, 256, 0, stream>>>(xT, Wih0h, Whh0h, b0, h1);
    // 4) layer 1: persistent recurrence, stores final h only
    lstm_layer<128, false><<<32, 256, 0, stream>>>(h1, Wih1h, Whh1h, b1, h2last);
    // 5) MLP head
    lstm_head<<<2, 256, 0, stream>>>(h2last, fc1w, fc1b, fc2w, fc2b, fc3w, fc3b,
                                     (float*)d_out);
}
